// OnnxCompatibleAttention_90417651516200
// MI455X (gfx1250) — matrix operand order repaired and checked by reading, not compiled, not run on hardware
//
#include <hip/hip_runtime.h>
#include <hip/hip_bf16.h>

// ---------------------------------------------------------------------------
// MHA for MI455X (gfx1250): bf16 WMMA everywhere. Both the projection GEMMs
// and the flash attention stage their tiles into LDS with
// global_load_async_to_lds_b128 (ASYNCcnt, double-buffered) so prefetch can't
// be sunk by the scheduler; compute consumes ds_load_b128 fragments.
// Attention is computed transposed so softmax reduces in-lane and exp'd S^T
// repacks directly into the P^T B-fragment.
// ---------------------------------------------------------------------------

typedef __bf16 bf16_t;
typedef __attribute__((ext_vector_type(16))) __bf16 v16bf;
typedef __attribute__((ext_vector_type(8)))  __bf16 v8bf;
typedef __attribute__((ext_vector_type(8)))  float  v8f;

#define DIMN   1024
#define HEADS  16
#define HDIM   64
#define BATCH  2
#define SEQ    2048
#define MTOT   (BATCH * SEQ)   // 4096 rows for the projection GEMMs

__device__ __forceinline__ bf16_t f2bf(float x) {
  unsigned u = __builtin_bit_cast(unsigned, x);
  unsigned r = (u + 0x7FFFu + ((u >> 16) & 1u)) >> 16;   // round-to-nearest-even
  unsigned short s = (unsigned short)r;
  return __builtin_bit_cast(bf16_t, s);
}

__device__ __forceinline__ v8f wmma_bf16(v16bf a, v16bf b, v8f c) {
  // (neg_a, A, neg_b, B, c_mod, C, reuse_a, reuse_b)
  return __builtin_amdgcn_wmma_f32_16x16x32_bf16(false, a, false, b, (short)0, c,
                                                 false, false);
}

// A-operand 16x32 bf16 fragment. Lane m(<16): row m, elems0..15 = K kc+0..15;
// lanes 16..31 get +16. Two 16B loads.
__device__ __forceinline__ v16bf load_a(const bf16_t* __restrict__ base, int ld,
                                        int kc, int lane) {
  const bf16_t* p = base + (size_t)(lane & 15) * ld + kc + ((lane >> 4) * 16);
  v8bf lo = *(const v8bf*)p;
  v8bf hi = *(const v8bf*)(p + 8);
  v16bf r;
#pragma unroll
  for (int i = 0; i < 8; ++i) { r[i] = lo[i]; r[i + 8] = hi[i]; }
  return r;
}

// B-operand 32x16 bf16 fragment. Lane n(<16): column n, K = kc+0..15 packed;
// lanes 16..31 hold K = kc+16..31. One 32B load per lane.
__device__ __forceinline__ v16bf load_b(const bf16_t* __restrict__ base, int ld,
                                        int kc, int lane) {
  const bf16_t* p = base + (size_t)(lane & 15) * ld + kc + ((lane >> 4) * 16);
  return *(const v16bf*)p;
}

// Async copy 16B global -> LDS (tracked by ASYNCcnt; no VGPR destination).
__device__ __forceinline__ void async_g2l_b128(void* lds_ptr, const void* gptr) {
  unsigned lds = (unsigned)(unsigned long long)lds_ptr;          // low 32 = LDS offset
  unsigned long long src = (unsigned long long)gptr;
  asm volatile("global_load_async_to_lds_b128 %0, %1, off"
               :: "v"(lds), "v"(src) : "memory");
}

__device__ __forceinline__ void wait_async0() {
  asm volatile("s_wait_asynccnt 0" ::: "memory");
}

// --------------------------- fp32 -> bf16 copy ------------------------------
__global__ void cvt_bf16_kernel(const float* __restrict__ src,
                                bf16_t* __restrict__ dst, int n) {
  int i = blockIdx.x * blockDim.x + threadIdx.x;
  int stride = gridDim.x * blockDim.x;
  for (; i < n; i += stride) dst[i] = f2bf(src[i]);
}

// --------------------------- QKV projections --------------------------------
// y[m,n] = sum_k x[m,k] * W[n,k] + bias[n],  m = b*SEQ+s, n = h*64+d
// Q,K stored bf16 as [B,H,S,D]; V stored transposed bf16 as [B,H,D,S].
// Block = 8 waves: tile M=128 x N=64, k staged 32 at a time through LDS.
__global__ void __launch_bounds__(256)
qkv_proj_kernel(const bf16_t* __restrict__ xb,
                const bf16_t* __restrict__ wq, const bf16_t* __restrict__ wk,
                const bf16_t* __restrict__ wv,
                const float* __restrict__ bq, const float* __restrict__ bk,
                const float* __restrict__ bv,
                bf16_t* __restrict__ qo, bf16_t* __restrict__ ko,
                bf16_t* __restrict__ vt) {
  __shared__ __align__(16) bf16_t At[2][128 * 32];   // [m][k]
  __shared__ __align__(16) bf16_t Wt[2][64 * 32];    // [n][k]

  const int tid  = threadIdx.x;
  const int lane = tid & 31;
  const int wave = tid >> 5;
  const int mblk = blockIdx.x * 128;
  const int m0 = mblk + wave * 16;
  const int n0 = blockIdx.y * 64;
  const int which = blockIdx.z;

  const bf16_t* W   = (which == 0) ? wq : (which == 1) ? wk : wv;
  const float* bias = (which == 0) ? bq : (which == 1) ? bk : bv;

  // staging coordinates: A tile 128x32 = 512 b128 (2/thread), W tile 64x32 = 256 (1/thread)
  const int ar0 = tid >> 2,          ac0 = (tid & 3) * 8;
  const int ar1 = (tid + 256) >> 2,  ac1 = (tid & 3) * 8;
  const int wr  = tid >> 2,          wc  = (tid & 3) * 8;

  auto stage = [&](int buf, int kc) {
    async_g2l_b128(&At[buf][ar0 * 32 + ac0],
                   xb + (size_t)(mblk + ar0) * DIMN + kc + ac0);
    async_g2l_b128(&At[buf][ar1 * 32 + ac1],
                   xb + (size_t)(mblk + ar1) * DIMN + kc + ac1);
    async_g2l_b128(&Wt[buf][wr * 32 + wc],
                   W + (size_t)(n0 + wr) * DIMN + kc + wc);
  };

  v8f acc[4] = {};
  int buf = 0;
  stage(buf, 0);

  for (int kc = 0; kc < DIMN; kc += 32) {
    wait_async0();
    __syncthreads();
    if (kc + 32 < DIMN) stage(buf ^ 1, kc + 32);

    v16bf a = load_a(&At[buf][wave * 16 * 32], 32, 0, lane);
#pragma unroll
    for (int nt = 0; nt < 4; ++nt) {
      v16bf b = load_b(&Wt[buf][nt * 16 * 32], 32, 0, lane);
      acc[nt] = wmma_bf16(a, b, acc[nt]);
    }
    buf ^= 1;
  }

  const int half = lane >> 4;
  const int nlane = lane & 15;
#pragma unroll
  for (int nt = 0; nt < 4; ++nt) {
    const int n = n0 + nt * 16 + nlane;
    const float bi = bias[n];
    const int h = n >> 6, d = n & 63;
    if (which < 2) {
      bf16_t* out = (which == 0) ? qo : ko;
#pragma unroll
      for (int r = 0; r < 8; ++r) {
        const int m = m0 + half * 8 + r;
        const int bb = m >> 11, s = m & (SEQ - 1);
        out[((size_t)(bb * HEADS + h) * SEQ + s) * HDIM + d] =
            f2bf(acc[nt][r] + bi);
      }
    } else {
      const int m = m0 + half * 8;               // 8 consecutive s values
      const int bb = m >> 11, s = m & (SEQ - 1);
      v8bf vv;
#pragma unroll
      for (int r = 0; r < 8; ++r) vv[r] = f2bf(acc[nt][r] + bi);
      *(v8bf*)(vt + (size_t)((bb * HEADS + h) * HDIM + d) * SEQ + s) = vv;
    }
  }
}

// --------------------------- flash attention --------------------------------
// Block = 8 waves, all sharing one (b,h); wave w owns queries [q0, q0+16).
// Per 32-key step the whole block stages K (32x64) and V^T (64x32) bf16 tiles
// into LDS via async loads (double-buffered), then every wave builds its WMMA
// fragments from LDS. S^T = K * Q^T so softmax reduces in-lane + shfl_xor(16),
// and exp'd S^T repacks directly into the P^T B-fragment.
__global__ void __launch_bounds__(256)
attention_kernel(const bf16_t* __restrict__ q, const bf16_t* __restrict__ k,
                 const bf16_t* __restrict__ vt, bf16_t* __restrict__ ctx) {
  __shared__ __align__(16) bf16_t Kt[2][32 * HDIM];   // [key][d]
  __shared__ __align__(16) bf16_t Vt[2][HDIM * 32];   // [d][key]

  const int tid  = threadIdx.x;
  const int lane = tid & 31;
  const int wave = tid >> 5;
  const int q0 = (blockIdx.x * 8 + wave) * 16;
  const int bh = blockIdx.y;                      // 0..31
  const int bb = bh >> 4, h = bh & 15;
  const int half = lane >> 4;

  const bf16_t* Q  = q  + (size_t)bh * SEQ * HDIM;
  const bf16_t* K  = k  + (size_t)bh * SEQ * HDIM;
  const bf16_t* VT = vt + (size_t)bh * HDIM * SEQ;

  // Q^T as B-operand (32(d) x 16(q)), two d-chunks held for the whole loop.
  const v16bf qb0 = load_b(Q + (size_t)q0 * HDIM, HDIM, 0,  lane);
  const v16bf qb1 = load_b(Q + (size_t)q0 * HDIM, HDIM, 32, lane);

  // staging coordinates (256 threads; one b128 per tile per thread)
  const int krow = tid >> 3, kcol = (tid & 7) * 8;   // K tile: 32 x 64
  const int vrow = tid >> 2, vcol = (tid & 3) * 8;   // V^T tile: 64 x 32

  auto stage = [&](int buf, int kb) {
    async_g2l_b128(&Kt[buf][krow * HDIM + kcol],
                   K + (size_t)(kb + krow) * HDIM + kcol);
    async_g2l_b128(&Vt[buf][vrow * 32 + vcol],
                   VT + (size_t)vrow * SEQ + kb + vcol);
  };

  v8f o[4] = {};                 // ctx^T accumulators: 4 d-tiles x (16x16)
  float m_run = -1e30f;
  float l_run = 0.0f;
  const float scale = 0.125f;    // HEAD_DIM^-0.5

  int buf = 0;
  stage(buf, 0);

  for (int kb = 0; kb < SEQ; kb += 32) {
    wait_async0();               // this wave's async writes to LDS landed
    __syncthreads();             // everyone's landed; prev buffer fully read
    if (kb + 32 < SEQ) stage(buf ^ 1, kb + 32);

    const bf16_t* Kc = &Kt[buf][0];
    const bf16_t* Vc = &Vt[buf][0];

    v8f s0 = {}, s1 = {};        // S^T: keys kb..+15 / kb+16..+31 (M), q (N)
    {
      v16bf ka = load_a(Kc, HDIM, 0, lane);
      s0 = wmma_bf16(ka, qb0, s0);
      ka = load_a(Kc, HDIM, 32, lane);
      s0 = wmma_bf16(ka, qb1, s0);
      ka = load_a(Kc + 16 * HDIM, HDIM, 0, lane);
      s1 = wmma_bf16(ka, qb0, s1);
      ka = load_a(Kc + 16 * HDIM, HDIM, 32, lane);
      s1 = wmma_bf16(ka, qb1, s1);
    }

    float mx = -1e30f;
#pragma unroll
    for (int e = 0; e < 8; ++e) {
      s0[e] *= scale; s1[e] *= scale;
      mx = fmaxf(mx, fmaxf(s0[e], s1[e]));
    }
    mx = fmaxf(mx, __shfl_xor(mx, 16, 32));       // merge lane halves (32 keys)
    const float m_new = fmaxf(m_run, mx);
    const float alpha = __expf(m_run - m_new);

    float p0[8], p1[8];
    float sum = 0.0f;
#pragma unroll
    for (int e = 0; e < 8; ++e) {
      p0[e] = __expf(s0[e] - m_new);
      p1[e] = __expf(s1[e] - m_new);
      sum += p0[e] + p1[e];
    }
    sum += __shfl_xor(sum, 16, 32);
    l_run = l_run * alpha + sum;
    m_run = m_new;

#pragma unroll
    for (int t = 0; t < 4; ++t)
#pragma unroll
      for (int e = 0; e < 8; ++e) o[t][e] *= alpha;

    // Repack P^T into a B-fragment (32(keys) x 16(q)):
    // lane<16 needs keys kb+0..15  -> own p0 | partner p0
    // lane>=16 needs keys kb+16..31 -> partner p1 | own p1
    v16bf pb;
#pragma unroll
    for (int e = 0; e < 8; ++e) {
      const float sh0 = __shfl_xor(p0[e], 16, 32);
      const float sh1 = __shfl_xor(p1[e], 16, 32);
      pb[e]     = f2bf((half == 0) ? p0[e] : sh1);
      pb[e + 8] = f2bf((half == 0) ? sh0 : p1[e]);
    }

    // ctx^T[d-tile, q] += V^T[d-tile, kb..kb+31] * P^T
#pragma unroll
    for (int t = 0; t < 4; ++t) {
      v16bf va = load_a(Vc + t * 16 * 32, 32, 0, lane);
      o[t] = wmma_bf16(va, pb, o[t]);
    }
    buf ^= 1;
  }

  // Normalize and store ctx as [B,S,DIM] bf16 (per lane: fixed q, 8 consecutive d).
  const float inv_l = 1.0f / l_run;
  const int qg = q0 + (lane & 15);
  bf16_t* crow = ctx + ((size_t)(bb * SEQ + qg)) * DIMN + h * HDIM;
#pragma unroll
  for (int t = 0; t < 4; ++t) {
    v8bf vv;
#pragma unroll
    for (int e = 0; e < 8; ++e) vv[e] = f2bf(o[t][e] * inv_l);
    *(v8bf*)(crow + t * 16 + half * 8) = vv;
  }
}

// --------------------------- output projection ------------------------------
// Same LDS-staged GEMM as qkv_proj, fp32 output with bias.
__global__ void __launch_bounds__(256)
out_proj_kernel(const bf16_t* __restrict__ cb, const bf16_t* __restrict__ wo,
                const float* __restrict__ bo, float* __restrict__ out) {
  __shared__ __align__(16) bf16_t At[2][128 * 32];   // [m][k]
  __shared__ __align__(16) bf16_t Wt[2][64 * 32];    // [n][k]

  const int tid  = threadIdx.x;
  const int lane = tid & 31;
  const int wave = tid >> 5;
  const int mblk = blockIdx.x * 128;
  const int m0 = mblk + wave * 16;
  const int n0 = blockIdx.y * 64;

  const int ar0 = tid >> 2,          ac0 = (tid & 3) * 8;
  const int ar1 = (tid + 256) >> 2,  ac1 = (tid & 3) * 8;
  const int wr  = tid >> 2,          wc  = (tid & 3) * 8;

  auto stage = [&](int buf, int kc) {
    async_g2l_b128(&At[buf][ar0 * 32 + ac0],
                   cb + (size_t)(mblk + ar0) * DIMN + kc + ac0);
    async_g2l_b128(&At[buf][ar1 * 32 + ac1],
                   cb + (size_t)(mblk + ar1) * DIMN + kc + ac1);
    async_g2l_b128(&Wt[buf][wr * 32 + wc],
                   wo + (size_t)(n0 + wr) * DIMN + kc + wc);
  };

  v8f acc[4] = {};
  int buf = 0;
  stage(buf, 0);

  for (int kc = 0; kc < DIMN; kc += 32) {
    wait_async0();
    __syncthreads();
    if (kc + 32 < DIMN) stage(buf ^ 1, kc + 32);

    v16bf a = load_a(&At[buf][wave * 16 * 32], 32, 0, lane);
#pragma unroll
    for (int nt = 0; nt < 4; ++nt) {
      v16bf b = load_b(&Wt[buf][nt * 16 * 32], 32, 0, lane);
      acc[nt] = wmma_bf16(a, b, acc[nt]);
    }
    buf ^= 1;
  }

  const int half = lane >> 4;
  const int nlane = lane & 15;
#pragma unroll
  for (int nt = 0; nt < 4; ++nt) {
    const int n = n0 + nt * 16 + nlane;
    const float bi = bo[n];
#pragma unroll
    for (int r = 0; r < 8; ++r) {
      const int m = m0 + half * 8 + r;
      out[(size_t)m * DIMN + n] = acc[nt][r] + bi;
    }
  }
}

// ---------------------------------------------------------------------------
extern "C" void kernel_launch(void* const* d_in, const int* in_sizes, int n_in,
                              void* d_out, int out_size, void* d_ws,
                              size_t ws_size, hipStream_t stream) {
  (void)in_sizes; (void)n_in; (void)out_size; (void)ws_size;

  const float* x  = (const float*)d_in[0];
  const float* Wq = (const float*)d_in[1];
  const float* bq = (const float*)d_in[2];
  const float* Wk = (const float*)d_in[3];
  const float* bk = (const float*)d_in[4];
  const float* Wv = (const float*)d_in[5];
  const float* bv = (const float*)d_in[6];
  const float* Wo = (const float*)d_in[7];
  const float* bo = (const float*)d_in[8];
  float* out = (float*)d_out;

  // Workspace carve-out (bf16), 256B aligned; total ~48 MiB.
  char* p = (char*)d_ws;
  auto carve = [&](size_t bytes) {
    void* r = (void*)p;
    p += (bytes + 255) & ~(size_t)255;
    return r;
  };
  bf16_t* xb   = (bf16_t*)carve((size_t)MTOT * DIMN * 2);
  bf16_t* wqb  = (bf16_t*)carve((size_t)DIMN * DIMN * 2);
  bf16_t* wkb  = (bf16_t*)carve((size_t)DIMN * DIMN * 2);
  bf16_t* wvb  = (bf16_t*)carve((size_t)DIMN * DIMN * 2);
  bf16_t* wob  = (bf16_t*)carve((size_t)DIMN * DIMN * 2);
  bf16_t* qb   = (bf16_t*)carve((size_t)MTOT * DIMN * 2);   // [B,H,S,D]
  bf16_t* kbuf = (bf16_t*)carve((size_t)MTOT * DIMN * 2);   // [B,H,S,D]
  bf16_t* vtb  = (bf16_t*)carve((size_t)MTOT * DIMN * 2);   // [B,H,D,S]
  bf16_t* ctxb = (bf16_t*)carve((size_t)MTOT * DIMN * 2);   // [B,S,DIM]

  // fp32 -> bf16 conversions
  cvt_bf16_kernel<<<1024, 256, 0, stream>>>(x,  xb,  MTOT * DIMN);
  cvt_bf16_kernel<<<512,  256, 0, stream>>>(Wq, wqb, DIMN * DIMN);
  cvt_bf16_kernel<<<512,  256, 0, stream>>>(Wk, wkb, DIMN * DIMN);
  cvt_bf16_kernel<<<512,  256, 0, stream>>>(Wv, wvb, DIMN * DIMN);
  cvt_bf16_kernel<<<512,  256, 0, stream>>>(Wo, wob, DIMN * DIMN);

  // QKV projections: grid = (M/128, N/64, 3 {q,k,v}), 8 waves/block
  qkv_proj_kernel<<<dim3(MTOT / 128, DIMN / 64, 3), 256, 0, stream>>>(
      xb, wqb, wkb, wvb, bq, bk, bv, qb, kbuf, vtb);

  // Flash attention: 8 waves/block = one (b,h), 8 query tiles, LDS-staged K/V
  attention_kernel<<<dim3((SEQ / 16) / 8, BATCH * HEADS), 256, 0, stream>>>(
      qb, kbuf, vtb, ctxb);

  // Output projection (fp32 out)
  out_proj_kernel<<<dim3(MTOT / 128, DIMN / 64), 256, 0, stream>>>(
      ctxb, wob, bo, out);
}
